// LstmAeUncertaintyModel_48636209660305
// MI455X (gfx1250) — compile-verified, matrix operand order checked by reading, not run on hardware
//
#include <hip/hip_runtime.h>
#include <hip/hip_bf16.h>

// ---------------------------------------------------------------------------
// LSTM autoencoder for MI455X (gfx1250): bf16 WMMA (16x16x32) with f32 acc.
// Per-step fused kernel: [x|h] A-panel staged to LDS once per block (8 waves
// share the batch tile), B tiles double-buffered (load k+1 while computing k),
// gate nonlinearity + c/h update fused in-registers via the C/D VGPR layout.
// ---------------------------------------------------------------------------

#define I_DIM 32
#define H_DIM 512
#define B_DIM 256
#define S_DIM 512
#define T_DIM 64

#define KC_LSTM 17            // (I + H) / 32 k-chunks ; chunk 0 == x (I=32)
#define KC_LIN  16            // H / 32
#define NT_LSTM 128           // 4H / 16 output column tiles
#define NT_LIN  2             // I / 16
#define GSTRIDE (32 * KC_LSTM * 512)  // packed-element stride between gates
#define A_STRIDE 552          // LDS A-panel row stride (bf16 elems), padded:
                              // 552*2B = 1104B -> m*276 dwords, m*20 mod 64
                              // distinct for m=0..15 (no bank-group repeats)

typedef __attribute__((ext_vector_type(16))) __bf16 v16bf;
typedef __attribute__((ext_vector_type(8)))  __bf16 v8bf;
typedef __attribute__((ext_vector_type(8)))  float  v8f;

__device__ __forceinline__ float sigmoidf_(float x) {
  return 1.0f / (1.0f + __expf(-x));
}

// ---------------- prep kernels -------------------------------------------

__global__ void k_cvt_bf16(const float* __restrict__ src,
                           __bf16* __restrict__ dst, int n) {
  int i = blockIdx.x * blockDim.x + threadIdx.x;
  if (i < n) dst[i] = (__bf16)src[i];
}

__global__ void k_init_state(__bf16* __restrict__ h0, float* __restrict__ c,
                             int n) {
  int i = blockIdx.x * blockDim.x + threadIdx.x;
  if (i < n) { h0[i] = (__bf16)0.0f; c[i] = 0.0f; }
}

__global__ void k_init_xdec(const float* __restrict__ seq,
                            __bf16* __restrict__ xdec) {
  int i = blockIdx.x * blockDim.x + threadIdx.x;
  if (i < B_DIM * I_DIM) {
    int b = i / I_DIM, f = i % I_DIM;
    xdec[i] = (__bf16)seq[(size_t)b * S_DIM * I_DIM +
                          (size_t)(S_DIM - 1) * I_DIM + f];
  }
}

// Pack [Wih | Whh] (row-major 4H x I and 4H x H) into per-lane WMMA-B layout:
// element id = ((ntile*KC + kc)*32 + lane)*16 + e, value = Wcat[n][k] where
// n = ntile*16 + (lane&15), k = kc*32 + (e&7) + ((e>>3)<<4) + ((lane>>4)<<3).
__global__ void k_pack_lstm(const float* __restrict__ Wih,
                            const float* __restrict__ Whh,
                            __bf16* __restrict__ pB) {
  int id = blockIdx.x * blockDim.x + threadIdx.x;
  if (id >= NT_LSTM * KC_LSTM * 512) return;
  int e     = id & 15;
  int lane  = (id >> 4) & 31;
  int kc    = (id >> 9) % KC_LSTM;
  int ntile = (id >> 9) / KC_LSTM;
  int n     = ntile * 16 + (lane & 15);
  int koff  = (e & 7) + ((e >> 3) << 4) + ((lane >> 4) << 3);
  int k     = kc * 32 + koff;
  float v   = (k < I_DIM) ? Wih[(size_t)n * I_DIM + k]
                          : Whh[(size_t)n * H_DIM + (k - I_DIM)];
  pB[id] = (__bf16)v;
}

__global__ void k_pack_lin(const float* __restrict__ W,
                           __bf16* __restrict__ pB) {
  int id = blockIdx.x * blockDim.x + threadIdx.x;
  if (id >= NT_LIN * KC_LIN * 512) return;
  int e     = id & 15;
  int lane  = (id >> 4) & 31;
  int kc    = (id >> 9) % KC_LIN;
  int ntile = (id >> 9) / KC_LIN;
  int n     = ntile * 16 + (lane & 15);
  int koff  = (e & 7) + ((e >> 3) << 4) + ((lane >> 4) << 3);
  int k     = kc * 32 + koff;
  pB[id] = (__bf16)W[(size_t)n * H_DIM + k];
}

// ---------------- fused LSTM step ----------------------------------------
// Grid: 64 blocks x 256 threads (8 waves).  All 8 waves of a block share one
// batch tile (bt = blockIdx>>2) and cover 8 hidden tiles.  The 16x544 A panel
// is staged into LDS once, then each wave runs a double-buffered WMMA loop.
__global__ void __launch_bounds__(256)
k_lstm_step(const __bf16* __restrict__ xsrc, int x_stride,
            const __bf16* __restrict__ pB, const float* __restrict__ bias,
            const __bf16* __restrict__ h_in, float* __restrict__ c,
            __bf16* __restrict__ h_out) {
  __shared__ __bf16 As[16 * A_STRIDE];

  const int lane = threadIdx.x & 31;
  const int wid  = (blockIdx.x << 3) + (threadIdx.x >> 5);
  const int bt   = wid >> 5;         // batch tile  0..15 (uniform per block)
  const int ht   = wid & 31;         // hidden tile 0..31
  const int m    = lane & 15;
  const int half = lane >> 4;

  // ---- stage A panel: cols [0,32) = x_t, cols [32,544) = h_{t-1} ----------
  for (int ch = threadIdx.x; ch < 1024; ch += 256) {      // h: 16 rows x 64
    const int r = ch >> 6, off = (ch & 63) << 3;
    *(v8bf*)&As[r * A_STRIDE + 32 + off] =
        *(const v8bf*)&h_in[(size_t)(bt * 16 + r) * H_DIM + off];
  }
  if (threadIdx.x < 64) {                                 // x: 16 rows x 4
    const int r = threadIdx.x >> 2, off = (threadIdx.x & 3) << 3;
    *(v8bf*)&As[r * A_STRIDE + off] =
        *(const v8bf*)&xsrc[(size_t)(bt * 16 + r) * x_stride + off];
  }
  __syncthreads();

  // ---- double-buffered WMMA main loop ------------------------------------
  v8f acc0 = {}, acc1 = {}, acc2 = {}, acc3 = {};
  const int kbase = half * 8;  // per-lane K sub-offset within a chunk

  v16bf a, an, b0, b1, b2, b3;
  {  // preload chunk 0
    const __bf16* bp = pB + ((size_t)ht * KC_LSTM) * 512 + (size_t)lane * 16;
    b0 = __builtin_nontemporal_load((const v16bf*)(bp));
    b1 = __builtin_nontemporal_load((const v16bf*)(bp + GSTRIDE));
    b2 = __builtin_nontemporal_load((const v16bf*)(bp + 2 * GSTRIDE));
    b3 = __builtin_nontemporal_load((const v16bf*)(bp + 3 * GSTRIDE));
    const v8bf a0 = *(const v8bf*)&As[m * A_STRIDE + kbase];
    const v8bf a1 = *(const v8bf*)&As[m * A_STRIDE + kbase + 16];
#pragma unroll
    for (int i = 0; i < 8; ++i) { a[i] = a0[i]; a[i + 8] = a1[i]; }
  }

#pragma unroll 2
  for (int kc = 0; kc < KC_LSTM - 1; ++kc) {
    // issue next chunk's loads before this chunk's WMMAs
    const __bf16* np =
        pB + ((size_t)ht * KC_LSTM + kc + 1) * 512 + (size_t)lane * 16;
    v16bf n0 = __builtin_nontemporal_load((const v16bf*)(np));
    v16bf n1 = __builtin_nontemporal_load((const v16bf*)(np + GSTRIDE));
    v16bf n2 = __builtin_nontemporal_load((const v16bf*)(np + 2 * GSTRIDE));
    v16bf n3 = __builtin_nontemporal_load((const v16bf*)(np + 3 * GSTRIDE));
    const int k0 = (kc + 1) * 32 + kbase;
    const v8bf a0 = *(const v8bf*)&As[m * A_STRIDE + k0];
    const v8bf a1 = *(const v8bf*)&As[m * A_STRIDE + k0 + 16];
#pragma unroll
    for (int i = 0; i < 8; ++i) { an[i] = a0[i]; an[i + 8] = a1[i]; }

    acc0 = __builtin_amdgcn_wmma_f32_16x16x32_bf16(false, a, false, b0,
                                                   (short)0, acc0, false, false);
    acc1 = __builtin_amdgcn_wmma_f32_16x16x32_bf16(false, a, false, b1,
                                                   (short)0, acc1, false, false);
    acc2 = __builtin_amdgcn_wmma_f32_16x16x32_bf16(false, a, false, b2,
                                                   (short)0, acc2, false, false);
    acc3 = __builtin_amdgcn_wmma_f32_16x16x32_bf16(false, a, false, b3,
                                                   (short)0, acc3, false, false);
    a = an; b0 = n0; b1 = n1; b2 = n2; b3 = n3;
  }
  // last chunk
  acc0 = __builtin_amdgcn_wmma_f32_16x16x32_bf16(false, a, false, b0,
                                                 (short)0, acc0, false, false);
  acc1 = __builtin_amdgcn_wmma_f32_16x16x32_bf16(false, a, false, b1,
                                                 (short)0, acc1, false, false);
  acc2 = __builtin_amdgcn_wmma_f32_16x16x32_bf16(false, a, false, b2,
                                                 (short)0, acc2, false, false);
  acc3 = __builtin_amdgcn_wmma_f32_16x16x32_bf16(false, a, false, b3,
                                                 (short)0, acc3, false, false);

  // ---- fused nonlinear update.  C/D layout: col = lane&15, row = 8*half+e.
  const int   ncol = ht * 16 + (lane & 15);
  const float bi   = bias[ncol];
  const float bf_  = bias[H_DIM + ncol];
  const float bg   = bias[2 * H_DIM + ncol];
  const float bo   = bias[3 * H_DIM + ncol];
  const int   row0 = bt * 16 + half * 8;
#pragma unroll
  for (int e = 0; e < 8; ++e) {
    const size_t idx = (size_t)(row0 + e) * H_DIM + ncol;
    const float gi = acc0[e] + bi;
    const float gf = acc1[e] + bf_;
    const float gg = acc2[e] + bg;
    const float go = acc3[e] + bo;
    const float cn = sigmoidf_(gf) * c[idx] + sigmoidf_(gi) * tanhf(gg);
    const float hn = sigmoidf_(go) * tanhf(cn);
    c[idx]     = cn;
    h_out[idx] = (__bf16)hn;
  }
}

// ---------------- decoder readout: y = relu(h) @ lin_W^T + lin_b ----------
__global__ void __launch_bounds__(256)
k_linear_step(const __bf16* __restrict__ h, const __bf16* __restrict__ pB,
              const float* __restrict__ bias, float* __restrict__ yout,
              __bf16* __restrict__ xnext) {
  const int lane  = threadIdx.x & 31;
  const int wid   = (blockIdx.x << 3) + (threadIdx.x >> 5); // 0..31
  const int bt    = wid >> 1;
  const int nt    = wid & 1;
  const int m     = lane & 15;
  const int half  = lane >> 4;
  const int batch = bt * 16 + m;

  v8f acc = {};
#pragma unroll 2
  for (int kc = 0; kc < KC_LIN; ++kc) {
    const __bf16* hp = h + (size_t)batch * H_DIM + kc * 32 + half * 8;
    v8bf a0 = *(const v8bf*)(hp);
    v8bf a1 = *(const v8bf*)(hp + 16);
    v16bf a;
#pragma unroll
    for (int i = 0; i < 8; ++i) {  // fused ReLU on the A operand
      float f0 = (float)a0[i];
      float f1 = (float)a1[i];
      a[i]     = (__bf16)(f0 > 0.0f ? f0 : 0.0f);
      a[i + 8] = (__bf16)(f1 > 0.0f ? f1 : 0.0f);
    }
    const __bf16* bp =
        pB + ((size_t)nt * KC_LIN + kc) * 512 + (size_t)lane * 16;
    v16bf b = *(const v16bf*)bp;
    acc = __builtin_amdgcn_wmma_f32_16x16x32_bf16(false, a, false, b,
                                                  (short)0, acc, false, false);
  }

  const int   ncol = nt * 16 + (lane & 15);
  const float bb   = bias[ncol];
  const int   row0 = bt * 16 + half * 8;
#pragma unroll
  for (int e = 0; e < 8; ++e) {
    const int r = row0 + e;
    const float y = acc[e] + bb;
    yout[(size_t)r * (T_DIM * I_DIM) + ncol] = y;     // out[b, t, :]
    xnext[r * I_DIM + ncol] = (__bf16)y;              // decoder input t+1
  }
}

// ---------------- host driver --------------------------------------------

extern "C" void kernel_launch(void* const* d_in, const int* in_sizes, int n_in,
                              void* d_out, int out_size, void* d_ws,
                              size_t ws_size, hipStream_t stream) {
  (void)in_sizes; (void)n_in; (void)out_size; (void)ws_size;

  const float* seq     = (const float*)d_in[0];
  const float* enc_Wih = (const float*)d_in[1];
  const float* enc_Whh = (const float*)d_in[2];
  const float* enc_b   = (const float*)d_in[3];
  const float* dec_Wih = (const float*)d_in[4];
  const float* dec_Whh = (const float*)d_in[5];
  const float* dec_b   = (const float*)d_in[6];
  const float* lin_W   = (const float*)d_in[7];
  const float* lin_b   = (const float*)d_in[8];
  float* out = (float*)d_out;

  // workspace carve (~14 MB)
  char* p = (char*)d_ws;
  auto carve = [&](size_t bytes) -> char* {
    char* r = p;
    p += (bytes + 255) & ~(size_t)255;
    return r;
  };
  __bf16* seq_bf = (__bf16*)carve((size_t)B_DIM * S_DIM * I_DIM * 2);
  __bf16* pB_enc = (__bf16*)carve((size_t)NT_LSTM * KC_LSTM * 512 * 2);
  __bf16* pB_dec = (__bf16*)carve((size_t)NT_LSTM * KC_LSTM * 512 * 2);
  __bf16* pB_lin = (__bf16*)carve((size_t)NT_LIN * KC_LIN * 512 * 2);
  __bf16* hbuf0  = (__bf16*)carve((size_t)B_DIM * H_DIM * 2);
  __bf16* hbuf1  = (__bf16*)carve((size_t)B_DIM * H_DIM * 2);
  float*  cbuf   = (float*) carve((size_t)B_DIM * H_DIM * 4);
  __bf16* xdec   = (__bf16*)carve((size_t)B_DIM * I_DIM * 2);

  const int nseq = B_DIM * S_DIM * I_DIM;
  k_cvt_bf16<<<(nseq + 255) / 256, 256, 0, stream>>>(seq, seq_bf, nseq);

  const int nplstm = NT_LSTM * KC_LSTM * 512;
  k_pack_lstm<<<(nplstm + 255) / 256, 256, 0, stream>>>(enc_Wih, enc_Whh,
                                                        pB_enc);
  k_pack_lstm<<<(nplstm + 255) / 256, 256, 0, stream>>>(dec_Wih, dec_Whh,
                                                        pB_dec);
  const int nplin = NT_LIN * KC_LIN * 512;
  k_pack_lin<<<(nplin + 255) / 256, 256, 0, stream>>>(lin_W, pB_lin);

  const int nstate = B_DIM * H_DIM;
  k_init_state<<<(nstate + 255) / 256, 256, 0, stream>>>(hbuf0, cbuf, nstate);
  k_init_xdec<<<(B_DIM * I_DIM + 255) / 256, 256, 0, stream>>>(seq, xdec);

  __bf16* hb[2] = {hbuf0, hbuf1};

  // encoder scan: 512 dependent steps, h ping-pong
  for (int t = 0; t < S_DIM; ++t) {
    k_lstm_step<<<64, 256, 0, stream>>>(seq_bf + (size_t)t * I_DIM,
                                        S_DIM * I_DIM, pB_enc, enc_b,
                                        hb[t & 1], cbuf, hb[(t + 1) & 1]);
  }
  // decoder scan: 64 steps, each LSTM step then fused relu+linear readout
  for (int t = 0; t < T_DIM; ++t) {
    k_lstm_step<<<64, 256, 0, stream>>>(xdec, I_DIM, pB_dec, dec_b,
                                        hb[t & 1], cbuf, hb[(t + 1) & 1]);
    k_linear_step<<<4, 256, 0, stream>>>(hb[(t + 1) & 1], pB_lin, lin_b,
                                         out + (size_t)t * I_DIM, xdec);
  }
}
